// GCN_64381559767785
// MI455X (gfx1250) — matrix operand order repaired and checked by reading, not compiled, not run on hardware
//
#include <hip/hip_runtime.h>
#include <hip/hip_bf16.h>

// ---------------------------------------------------------------------------
// GCN forward: 3x GCNConv (GEMM + normalized edge scatter) + mean pool + head
// GEMMs use v_wmma_f32_16x16x32_f16 (wave32 CDNA5 WMMA).
// ---------------------------------------------------------------------------

typedef __attribute__((ext_vector_type(16))) _Float16 v16h;
typedef __attribute__((ext_vector_type(8)))  float    v8f;

#define N_NODES   50000
#define N_EDGES   800000
#define HID       64
#define NUM_GRAPHS 512
#define M_TILES   (N_NODES / 16)   // 3125, exact

// ---------------- WMMA fragment loaders ------------------------------------
// A fragment (16x32, f16), per ISA layout:
//   lanes 0-15 : M=lid, elems 0..7 -> K=kk+0..7,   elems 8..15 -> K=kk+16..23
//   lanes 16-31: M=lid, elems 0..7 -> K=kk+8..15,  elems 8..15 -> K=kk+24..31
template<bool RELU>
__device__ __forceinline__ v16h load_A_frag(const float* __restrict__ in,
                                            int mbase, int kk, int lid, int hf) {
    const float* p = in + (size_t)(mbase + lid) * HID + kk + hf * 8;
    float4 q0 = *reinterpret_cast<const float4*>(p);
    float4 q1 = *reinterpret_cast<const float4*>(p + 4);
    float4 q2 = *reinterpret_cast<const float4*>(p + 16);
    float4 q3 = *reinterpret_cast<const float4*>(p + 20);
    float t[16] = {q0.x, q0.y, q0.z, q0.w, q1.x, q1.y, q1.z, q1.w,
                   q2.x, q2.y, q2.z, q2.w, q3.x, q3.y, q3.z, q3.w};
    v16h a;
#pragma unroll
    for (int j = 0; j < 16; ++j) {
        float v = t[j];
        if (RELU) v = fmaxf(v, 0.0f);
        a[j] = (_Float16)v;
    }
    return a;
}

// B fragment (32x16, f16): lanes 0-15 N=lid hold K=kk+0..7,kk+16..23,
// lanes 16-31 N=lid hold K=kk+8..15,kk+24..31.  W is row-major [K][N].
__device__ __forceinline__ v16h load_B_frag(const float* __restrict__ W,
                                            int nbase, int kk, int lid, int hf) {
    const float* p = W + (size_t)(kk + hf * 8) * HID + nbase + lid;
    v16h b;
#pragma unroll
    for (int j = 0; j < 16; ++j) b[j] = (_Float16)p[(size_t)(j < 8 ? j : j + 8) * HID];
    return b;
}

// ---------------- GEMM: out[N_NODES x 64] = relu?(in) @ W ------------------
// 8 waves / block. wave w: ntile = w&3 (B frags cached in regs),
// m-tiles = blockIdx.x*8 + (w>>2), step 2  (4 m-tiles per wave).
template<bool RELU>
__global__ __launch_bounds__(256)
void k_gemm(const float* __restrict__ in, const float* __restrict__ W,
            float* __restrict__ out) {
    const int w    = threadIdx.x >> 5;
    const int lane = threadIdx.x & 31;
    const int lid  = lane & 15;
    const int hf   = lane >> 4;
    const int nbase = (w & 3) * 16;

    v16h b0 = load_B_frag(W, nbase, 0,  lid, hf);
    v16h b1 = load_B_frag(W, nbase, 32, lid, hf);

    int mt_end = blockIdx.x * 8 + 8;
    if (mt_end > M_TILES) mt_end = M_TILES;
    for (int mt = blockIdx.x * 8 + (w >> 2); mt < mt_end; mt += 2) {
        const int mbase = mt * 16;
        v8f acc = {0.f, 0.f, 0.f, 0.f, 0.f, 0.f, 0.f, 0.f};
        v16h a0 = load_A_frag<RELU>(in, mbase, 0, lid, hf);
        acc = __builtin_amdgcn_wmma_f32_16x16x32_f16(false, a0, false, b0,
                                                     (short)0, acc, false, false);
        v16h a1 = load_A_frag<RELU>(in, mbase, 32, lid, hf);
        acc = __builtin_amdgcn_wmma_f32_16x16x32_f16(false, a1, false, b1,
                                                     (short)0, acc, false, false);
        // C layout: lanes 0-15 -> M = mbase + r, lanes 16-31 -> M = mbase+8+r
        float* op = out + (size_t)(mbase + hf * 8) * HID + nbase + lid;
#pragma unroll
        for (int r = 0; r < 8; ++r) op[(size_t)r * HID] = acc[r];
    }
}

// ---------------- degree / norm --------------------------------------------
__global__ void k_fill(float* __restrict__ p, int n, float v) {
    int i = blockIdx.x * blockDim.x + threadIdx.x;
    if (i < n) p[i] = v;
}

__global__ void k_count_deg(const int* __restrict__ col, int e_cnt,
                            float* __restrict__ deg) {
    int e = blockIdx.x * blockDim.x + threadIdx.x;
    if (e < e_cnt) atomicAdd(&deg[col[e]], 1.0f);
}

__global__ void k_dinv(const float* __restrict__ deg, float* __restrict__ dinv, int n) {
    int i = blockIdx.x * blockDim.x + threadIdx.x;
    if (i < n) dinv[i] = rsqrtf(deg[i]);   // deg >= 1 (self-loop folded in)
}

// ---------------- self-loop + bias init ------------------------------------
// out[i][f] = b[f] + dinv[i]^2 * h[i][f]
__global__ void k_init_out(const float* __restrict__ h, const float* __restrict__ dinv,
                           const float* __restrict__ b, float* __restrict__ out) {
    int t = blockIdx.x * blockDim.x + threadIdx.x;
    if (t >= N_NODES * HID) return;
    int i = t >> 6, f = t & 63;
    float di = dinv[i];
    out[t] = b[f] + di * di * h[t];
}

// ---------------- edge scatter: one wave per edge --------------------------
__global__ __launch_bounds__(256)
void k_scatter(const float* __restrict__ h, const float* __restrict__ dinv,
               const int* __restrict__ ei, float* __restrict__ out) {
    int wid  = (int)((blockIdx.x * blockDim.x + threadIdx.x) >> 5);
    int lane = threadIdx.x & 31;
    if (wid >= N_EDGES) return;
    int r = ei[wid];              // source
    int c = ei[N_EDGES + wid];    // target
    float nrm = dinv[r] * dinv[c];
    const float* hp = h + (size_t)r * HID;
    float* op = out + (size_t)c * HID;
    atomicAdd(&op[lane],      nrm * hp[lane]);
    atomicAdd(&op[lane + 32], nrm * hp[lane + 32]);
}

// ---------------- pooling + head -------------------------------------------
__global__ void k_pool_sum(const float* __restrict__ h, const int* __restrict__ batch,
                           float* __restrict__ psum) {
    int t = blockIdx.x * blockDim.x + threadIdx.x;
    if (t >= N_NODES * HID) return;
    int i = t >> 6, f = t & 63;
    atomicAdd(&psum[(size_t)batch[i] * HID + f], h[t]);
}

__global__ void k_pool_cnt(const int* __restrict__ batch, float* __restrict__ pcnt) {
    int i = blockIdx.x * blockDim.x + threadIdx.x;
    if (i < N_NODES) atomicAdd(&pcnt[batch[i]], 1.0f);
}

__global__ void k_final(const float* __restrict__ psum, const float* __restrict__ pcnt,
                        const float* __restrict__ linW, const float* __restrict__ linb,
                        float* __restrict__ out) {
    int g = blockIdx.x * blockDim.x + threadIdx.x;
    if (g >= NUM_GRAPHS) return;
    float inv = 1.0f / fmaxf(pcnt[g], 1.0f);
    float s = 0.0f;
#pragma unroll 8
    for (int f = 0; f < HID; ++f) s += psum[(size_t)g * HID + f] * linW[f];
    out[g] = s * inv + linb[0];
}

// ---------------------------------------------------------------------------
extern "C" void kernel_launch(void* const* d_in, const int* in_sizes, int n_in,
                              void* d_out, int out_size, void* d_ws, size_t ws_size,
                              hipStream_t stream) {
    const float* x    = (const float*)d_in[0];
    const float* W1   = (const float*)d_in[1];
    const float* b1   = (const float*)d_in[2];
    const float* W2   = (const float*)d_in[3];
    const float* b2   = (const float*)d_in[4];
    const float* W3   = (const float*)d_in[5];
    const float* b3   = (const float*)d_in[6];
    const float* linW = (const float*)d_in[7];
    const float* linb = (const float*)d_in[8];
    const int*   ei   = (const int*)d_in[9];    // [2, E]
    const int*   batch= (const int*)d_in[10];   // [N]
    float* out = (float*)d_out;

    // workspace carve-up (256B aligned slabs)
    char* base = (char*)d_ws;
    size_t off = 0;
    auto take = [&](size_t bytes) {
        char* p = base + off;
        off += (bytes + 255) & ~(size_t)255;
        return p;
    };
    float* deg  = (float*)take((size_t)N_NODES * 4);
    float* dinv = (float*)take((size_t)N_NODES * 4);
    float* h    = (float*)take((size_t)N_NODES * HID * 4);
    float* bufA = (float*)take((size_t)N_NODES * HID * 4);
    float* bufB = (float*)take((size_t)N_NODES * HID * 4);
    float* psum = (float*)take((size_t)NUM_GRAPHS * HID * 4);
    float* pcnt = (float*)take((size_t)NUM_GRAPHS * 4);

    const int TB = 256;
    const int gN   = (N_NODES + TB - 1) / TB;            // 196
    const int gE   = (N_EDGES + TB - 1) / TB;            // 3125
    const int gNF  = (N_NODES * HID + TB - 1) / TB;      // 12500
    const int gSc  = N_EDGES / 8;                        // 100000 (wave/edge)
    const int gMM  = (M_TILES + 7) / 8;                  // 391

    // degrees (init to 1.0 = self-loop), symmetric norm
    k_fill<<<gN, TB, 0, stream>>>(deg, N_NODES, 1.0f);
    k_count_deg<<<gE, TB, 0, stream>>>(ei + N_EDGES, N_EDGES, deg);
    k_dinv<<<gN, TB, 0, stream>>>(deg, dinv, N_NODES);

    // layer 1: x -> bufA
    k_gemm<false><<<gMM, TB, 0, stream>>>(x, W1, h);
    k_init_out<<<gNF, TB, 0, stream>>>(h, dinv, b1, bufA);
    k_scatter<<<gSc, TB, 0, stream>>>(h, dinv, ei, bufA);

    // layer 2: relu(bufA) -> bufB
    k_gemm<true><<<gMM, TB, 0, stream>>>(bufA, W2, h);
    k_init_out<<<gNF, TB, 0, stream>>>(h, dinv, b2, bufB);
    k_scatter<<<gSc, TB, 0, stream>>>(h, dinv, ei, bufB);

    // layer 3: relu(bufB) -> bufA (no output relu)
    k_gemm<true><<<gMM, TB, 0, stream>>>(bufB, W3, h);
    k_init_out<<<gNF, TB, 0, stream>>>(h, dinv, b3, bufA);
    k_scatter<<<gSc, TB, 0, stream>>>(h, dinv, ei, bufA);

    // mean pool + linear head
    k_fill<<<(NUM_GRAPHS * HID + TB - 1) / TB, TB, 0, stream>>>(psum, NUM_GRAPHS * HID, 0.0f);
    k_fill<<<(NUM_GRAPHS + TB - 1) / TB, TB, 0, stream>>>(pcnt, NUM_GRAPHS, 0.0f);
    k_pool_sum<<<gNF, TB, 0, stream>>>(bufA, batch, psum);
    k_pool_cnt<<<gN, TB, 0, stream>>>(batch, pcnt);
    k_final<<<(NUM_GRAPHS + TB - 1) / TB, TB, 0, stream>>>(psum, pcnt, linW, linb, out);
}
